// AttentionBlock_16097537425834
// MI455X (gfx1250) — matrix operand order repaired and checked by reading, not compiled, not run on hardware
//
#include <hip/hip_runtime.h>
#include <hip/hip_bf16.h>

#define B_ 4
#define C_ 512
#define N_ 4096
#define G_ 8
#define CG_ 64
#define QK_SCALE 0.04419417382415922f   // 512^-0.5

typedef __attribute__((ext_vector_type(16))) __bf16 v16bf;
typedef __attribute__((ext_vector_type(8)))  float  v8f;

union FragAB {
  v16bf v;
  uint4 q[2];
  unsigned short h[16];
};
union FragC {
  v8f v;
  float f[8];
};

__device__ __forceinline__ unsigned short f2bf(float x) {
  unsigned u = __float_as_uint(x);
  u += 0x7FFFu + ((u >> 16) & 1u);      // round-to-nearest-even truncation
  return (unsigned short)(u >> 16);
}

// A-matrix (16xK tile, 16-bit): lane row = lane&15; elements 0..7 = k0+(hi?8:0)+0..7,
// elements 8..15 = same +16.  Source row-major [row][ld] in bf16 (ushort) units.
__device__ __forceinline__ void load_a(FragAB& f, const unsigned short* __restrict__ base,
                                       int ld, int row, int k0, int lane) {
  const unsigned short* p = base + (size_t)(row + (lane & 15)) * ld + (k0 + ((lane >> 4) << 3));
  f.q[0] = *(const uint4*)p;
  f.q[1] = *(const uint4*)(p + 16);
}
// Same A-fragment gather but from an LDS-resident tile (emits ds_load_b128).
__device__ __forceinline__ void load_a_lds(FragAB& f, const unsigned short* base,
                                           int ld, int row, int k0, int lane) {
  const unsigned short* p = base + (row + (lane & 15)) * ld + (k0 + ((lane >> 4) << 3));
  f.q[0] = *(const uint4*)p;
  f.q[1] = *(const uint4*)(p + 16);
}
// B-matrix (Kx16 tile, 16-bit): lane col = lane&15; elements 0..7 = k0+(hi?8:0)+0..7,
// elements 8..15 = same +16.
// Source stored "column-major" for B, i.e. [col][ld] rows contiguous in k.
__device__ __forceinline__ void load_b(FragAB& f, const unsigned short* __restrict__ base,
                                       int ld, int col, int k0, int lane) {
  const unsigned short* p = base + (size_t)(col + (lane & 15)) * ld + (k0 + ((lane >> 4) << 3));
  f.q[0] = *(const uint4*)p;
  f.q[1] = *(const uint4*)(p + 16);
}
__device__ __forceinline__ v8f wmma_bf16(const FragAB& a, const FragAB& b, v8f c) {
  return __builtin_amdgcn_wmma_f32_16x16x32_bf16(false, a.v, false, b.v, (short)0, c, false, false);
}

// ---------------------------------------------------------------------------
// f32 -> bf16 conversion (weights)
// ---------------------------------------------------------------------------
__global__ __launch_bounds__(256) void cvt_bf16(const float* __restrict__ in,
                                                unsigned short* __restrict__ out, int n) {
  int i = blockIdx.x * 256 + threadIdx.x;
  if (i < n) out[i] = f2bf(in[i]);
}

// ---------------------------------------------------------------------------
// GroupNorm statistics: one workgroup per (b, g); region is contiguous 1MB.
// ---------------------------------------------------------------------------
__global__ __launch_bounds__(256) void gn_stats(const float* __restrict__ x,
                                                float* __restrict__ stats) {
  __shared__ float rs[256], rq[256];
  int bg = blockIdx.x;                         // 0..31
  const float* p = x + (size_t)bg * (CG_ * N_);
  float s = 0.f, q = 0.f;
  for (int i = threadIdx.x; i < CG_ * N_; i += 256) {
    float v = p[i]; s += v; q += v * v;
  }
  rs[threadIdx.x] = s; rq[threadIdx.x] = q;
  __syncthreads();
  for (int off = 128; off > 0; off >>= 1) {
    if (threadIdx.x < (unsigned)off) {
      rs[threadIdx.x] += rs[threadIdx.x + off];
      rq[threadIdx.x] += rq[threadIdx.x + off];
    }
    __syncthreads();
  }
  if (threadIdx.x == 0) {
    float inv = 1.0f / (float)(CG_ * N_);
    float mean = rs[0] * inv;
    float var = rq[0] * inv - mean * mean;
    stats[bg * 2] = mean;
    stats[bg * 2 + 1] = rsqrtf(var + 1e-5f);
  }
}

// ---------------------------------------------------------------------------
// GroupNorm apply; writes h transposed [b][n][c] in bf16 (B-fragment layout).
// ---------------------------------------------------------------------------
__global__ __launch_bounds__(256) void gn_apply(const float* __restrict__ x,
                                                const float* __restrict__ gamma,
                                                const float* __restrict__ beta,
                                                const float* __restrict__ stats,
                                                unsigned short* __restrict__ ht) {
  int idx = blockIdx.x * 256 + threadIdx.x;     // < B*C*N = 8388608
  int n = idx % N_;
  int c = (idx / N_) % C_;
  int b = idx / (N_ * C_);
  int bg = b * G_ + (c >> 6);
  float mean = stats[bg * 2];
  float rstd = stats[bg * 2 + 1];
  float h = (x[idx] - mean) * rstd * gamma[c] + beta[c];
  ht[((size_t)b * N_ + n) * C_ + c] = f2bf(h);
}

// ---------------------------------------------------------------------------
// QKV GEMM: qkv[b][o][n] = w_qkv[o][:] . h[b][:][n] + bias.  M=1536,N=4096,K=512.
// 128x128 block, 8 waves each 32(row)x64(col) = 2x4 WMMA tiles.
// Routes outputs: q^T (scaled) [n][c], k^T [n][c], v [c][m].
// ---------------------------------------------------------------------------
__global__ __launch_bounds__(256) void qkv_gemm(const unsigned short* __restrict__ wq,
                                                const unsigned short* __restrict__ ht,
                                                const float* __restrict__ bqkv,
                                                unsigned short* __restrict__ qt,
                                                unsigned short* __restrict__ kt,
                                                unsigned short* __restrict__ vc) {
  int b = blockIdx.z;
  int o0 = blockIdx.y * 128;
  int n0 = blockIdx.x * 128;
  int tid = threadIdx.x, lane = tid & 31, wave = tid >> 5;
  int ln15 = lane & 15, hi8 = (lane & 16) ? 8 : 0;
  int orow = o0 + (wave & 3) * 32;
  int ncol = n0 + (wave >> 2) * 64;
  const unsigned short* hb = ht + (size_t)b * N_ * C_;

  FragC acc[2][4];
#pragma unroll
  for (int r = 0; r < 2; ++r)
#pragma unroll
    for (int c = 0; c < 4; ++c)
#pragma unroll
      for (int i = 0; i < 8; ++i) acc[r][c].f[i] = 0.f;

  for (int kk = 0; kk < C_; kk += 32) {
    FragAB a0, a1, bb[4];
    load_a(a0, wq, C_, orow, kk, lane);
    load_a(a1, wq, C_, orow + 16, kk, lane);
#pragma unroll
    for (int c = 0; c < 4; ++c) load_b(bb[c], hb, C_, ncol + c * 16, kk, lane);
    if (kk + 32 < C_)
      __builtin_prefetch(hb + (size_t)(ncol + ln15) * C_ + kk + 32, 0, 3);
#pragma unroll
    for (int c = 0; c < 4; ++c) {
      acc[0][c].v = wmma_bf16(a0, bb[c], acc[0][c].v);
      acc[1][c].v = wmma_bf16(a1, bb[c], acc[1][c].v);
    }
  }
#pragma unroll
  for (int r = 0; r < 2; ++r)
#pragma unroll
    for (int c = 0; c < 4; ++c) {
      int nc = ncol + c * 16 + ln15;
#pragma unroll
      for (int i = 0; i < 8; ++i) {
        int o = orow + r * 16 + hi8 + i;
        float val = acc[r][c].f[i] + bqkv[o];
        if (o < C_) {
          qt[((size_t)b * N_ + nc) * C_ + o] = f2bf(val * QK_SCALE);
        } else if (o < 2 * C_) {
          kt[((size_t)b * N_ + nc) * C_ + (o - C_)] = f2bf(val);
        } else {
          vc[((size_t)b * C_ + (o - 2 * C_)) * N_ + nc] = f2bf(val);
        }
      }
    }
}

// ---------------------------------------------------------------------------
// Flash attention: one WG per 32 query rows; chunks of 128 keys.
// Q block (32x512 bf16 = 32KB) is staged into LDS ONCE via CDNA5 async
// global->LDS copies (ASYNCcnt), then re-read 32x from LDS instead of global.
// Phase1: S=QK^T via WMMA -> LDS. Phase2: online softmax stats.
// Phase3: O = diag(alpha) O + P V via WMMA (accumulators in registers).
// ---------------------------------------------------------------------------
#define BR 32
#define BC 128
__global__ __launch_bounds__(256) void attn_kernel(const unsigned short* __restrict__ qt,
                                                   const unsigned short* __restrict__ kt,
                                                   const unsigned short* __restrict__ vc,
                                                   unsigned short* __restrict__ at) {
  __shared__ float S[BR * BC];
  __shared__ unsigned short Qs[BR * C_];   // 32KB staged Q block
  __shared__ float alphaS[BR];
  __shared__ float mS[BR];
  int b = blockIdx.y;
  int n0 = blockIdx.x * BR;
  int tid = threadIdx.x, lane = tid & 31, wave = tid >> 5;
  int ln15 = lane & 15, hi8 = (lane & 16) ? 8 : 0;
  int rt1 = wave & 1;            // phase-1 row tile (0..1)
  int cg  = (wave >> 1) * 32;    // phase-1 col base within chunk
  int c0w = wave * 64;           // phase-3 output-col base within C

  const unsigned short* qb = qt + (size_t)b * N_ * C_;
  const unsigned short* kb = kt + (size_t)b * N_ * C_;
  const unsigned short* vb = vc + (size_t)b * C_ * N_;

  // ---- Stage Q block (rows n0..n0+31 are contiguous 32KB in [n][c] layout) ----
  {
    const unsigned short* gsrc = qb + (size_t)n0 * C_;
    unsigned qs_base = (unsigned)(size_t)(void*)Qs;   // LDS byte offset (addr[31:0])
    unsigned off = (unsigned)tid * 16u;               // bytes
#pragma unroll
    for (int it = 0; it < 8; ++it) {                  // 8 * 256 * 16B = 32KB
      unsigned ldsa = qs_base + off;
      asm volatile("global_load_async_to_lds_b128 %0, %1, %2"
                   :: "v"(ldsa), "v"(off), "s"(gsrc)
                   : "memory");
      off += 256u * 16u;
    }
    asm volatile("s_wait_asynccnt 0x0" ::: "memory");
  }
  __syncthreads();

  FragC oacc[2][4];
#pragma unroll
  for (int r = 0; r < 2; ++r)
#pragma unroll
    for (int c = 0; c < 4; ++c)
#pragma unroll
      for (int i = 0; i < 8; ++i) oacc[r][c].f[i] = 0.f;

  float Mrun = -3.0e38f, Lrun = 0.f, alpha_r = 0.f;   // live in lanes tid<32

  for (int mc = 0; mc < N_; mc += BC) {
    // ---- Phase 1: S tile (rows rt1*16.., cols cg..cg+31) ----
    FragC sacc[2];
#pragma unroll
    for (int c = 0; c < 2; ++c)
#pragma unroll
      for (int i = 0; i < 8; ++i) sacc[c].f[i] = 0.f;

    for (int kk = 0; kk < C_; kk += 32) {
      FragAB aq, bk0, bk1;
      load_a_lds(aq, Qs, C_, rt1 * 16, kk, lane);
      load_b(bk0, kb, C_, mc + cg,      kk, lane);
      load_b(bk1, kb, C_, mc + cg + 16, kk, lane);
      sacc[0].v = wmma_bf16(aq, bk0, sacc[0].v);
      sacc[1].v = wmma_bf16(aq, bk1, sacc[1].v);
    }
#pragma unroll
    for (int c = 0; c < 2; ++c) {
      int colb = cg + c * 16 + ln15;
#pragma unroll
      for (int i = 0; i < 8; ++i)
        S[(rt1 * 16 + hi8 + i) * BC + colb] = sacc[c].f[i];
    }
    __syncthreads();

    // ---- Row max / rescale factor (one lane per row) ----
    if (tid < BR) {
      float mx = -3.0e38f;
      for (int j = 0; j < BC; ++j) mx = fmaxf(mx, S[tid * BC + j]);
      float Mnew = fmaxf(Mrun, mx);
      alpha_r = __expf(Mrun - Mnew);
      Mrun = Mnew;
      alphaS[tid] = alpha_r;
      mS[tid] = Mnew;
    }
    __syncthreads();

    // ---- P = exp(S - Mnew), in place ----
    for (int e = tid; e < BR * BC; e += 256) {
      int r = e >> 7;
      S[e] = __expf(S[e] - mS[r]);
    }
    __syncthreads();

    // ---- Row sums -> running L ----
    if (tid < BR) {
      float s = 0.f;
      for (int j = 0; j < BC; ++j) s += S[tid * BC + j];
      Lrun = alpha_r * Lrun + s;
    }

    // ---- Phase 3: rescale accumulators and O += P * V ----
#pragma unroll
    for (int r = 0; r < 2; ++r)
#pragma unroll
      for (int i = 0; i < 8; ++i) {
        float a = alphaS[r * 16 + hi8 + i];
#pragma unroll
        for (int c = 0; c < 4; ++c) oacc[r][c].f[i] *= a;
      }

    for (int km = 0; km < 4; ++km) {
      int kofs = km * 32;
      FragAB ap[2];
#pragma unroll
      for (int r = 0; r < 2; ++r) {
        const float* ps = &S[(r * 16 + ln15) * BC + kofs + hi8];
        float4 f0 = *(const float4*)(ps);
        float4 f1 = *(const float4*)(ps + 4);
        float4 f2 = *(const float4*)(ps + 16);
        float4 f3 = *(const float4*)(ps + 20);
        ap[r].h[0]  = f2bf(f0.x); ap[r].h[1]  = f2bf(f0.y);
        ap[r].h[2]  = f2bf(f0.z); ap[r].h[3]  = f2bf(f0.w);
        ap[r].h[4]  = f2bf(f1.x); ap[r].h[5]  = f2bf(f1.y);
        ap[r].h[6]  = f2bf(f1.z); ap[r].h[7]  = f2bf(f1.w);
        ap[r].h[8]  = f2bf(f2.x); ap[r].h[9]  = f2bf(f2.y);
        ap[r].h[10] = f2bf(f2.z); ap[r].h[11] = f2bf(f2.w);
        ap[r].h[12] = f2bf(f3.x); ap[r].h[13] = f2bf(f3.y);
        ap[r].h[14] = f2bf(f3.z); ap[r].h[15] = f2bf(f3.w);
      }
      FragAB bv[4];
#pragma unroll
      for (int c = 0; c < 4; ++c)
        load_b(bv[c], vb, N_, c0w + c * 16, mc + kofs, lane);
#pragma unroll
      for (int r = 0; r < 2; ++r)
#pragma unroll
        for (int c = 0; c < 4; ++c)
          oacc[r][c].v = wmma_bf16(ap[r], bv[c], oacc[r][c].v);
    }
    __syncthreads();
  }

  if (tid < BR) mS[tid] = 1.0f / Lrun;
  __syncthreads();

#pragma unroll
  for (int r = 0; r < 2; ++r)
#pragma unroll
    for (int c = 0; c < 4; ++c) {
      int col = c0w + c * 16 + ln15;
#pragma unroll
      for (int i = 0; i < 8; ++i) {
        int row = r * 16 + hi8 + i;
        float val = oacc[r][c].f[i] * mS[row];
        at[((size_t)b * N_ + n0 + row) * C_ + col] = f2bf(val);
      }
    }
}

// ---------------------------------------------------------------------------
// Output projection + bias + residual.  M=512, N=4096, K=512.
// ---------------------------------------------------------------------------
__global__ __launch_bounds__(256) void proj_gemm(const unsigned short* __restrict__ wp,
                                                 const unsigned short* __restrict__ at,
                                                 const float* __restrict__ bp,
                                                 const float* __restrict__ x,
                                                 float* __restrict__ out) {
  int b = blockIdx.z;
  int o0 = blockIdx.y * 128;
  int n0 = blockIdx.x * 128;
  int tid = threadIdx.x, lane = tid & 31, wave = tid >> 5;
  int ln15 = lane & 15, hi8 = (lane & 16) ? 8 : 0;
  int orow = o0 + (wave & 3) * 32;
  int ncol = n0 + (wave >> 2) * 64;
  const unsigned short* ab = at + (size_t)b * N_ * C_;

  FragC acc[2][4];
#pragma unroll
  for (int r = 0; r < 2; ++r)
#pragma unroll
    for (int c = 0; c < 4; ++c)
#pragma unroll
      for (int i = 0; i < 8; ++i) acc[r][c].f[i] = 0.f;

  for (int kk = 0; kk < C_; kk += 32) {
    FragAB a0, a1, bb[4];
    load_a(a0, wp, C_, orow, kk, lane);
    load_a(a1, wp, C_, orow + 16, kk, lane);
#pragma unroll
    for (int c = 0; c < 4; ++c) load_b(bb[c], ab, C_, ncol + c * 16, kk, lane);
    if (kk + 32 < C_)
      __builtin_prefetch(ab + (size_t)(ncol + ln15) * C_ + kk + 32, 0, 3);
#pragma unroll
    for (int c = 0; c < 4; ++c) {
      acc[0][c].v = wmma_bf16(a0, bb[c], acc[0][c].v);
      acc[1][c].v = wmma_bf16(a1, bb[c], acc[1][c].v);
    }
  }
#pragma unroll
  for (int r = 0; r < 2; ++r)
#pragma unroll
    for (int c = 0; c < 4; ++c) {
      int nc = ncol + c * 16 + ln15;
#pragma unroll
      for (int i = 0; i < 8; ++i) {
        int o = orow + r * 16 + hi8 + i;
        size_t oi = ((size_t)b * C_ + o) * N_ + nc;
        out[oi] = acc[r][c].f[i] + bp[o] + x[oi];
      }
    }
}

// ---------------------------------------------------------------------------
extern "C" void kernel_launch(void* const* d_in, const int* in_sizes, int n_in,
                              void* d_out, int out_size, void* d_ws, size_t ws_size,
                              hipStream_t stream) {
  const float* x      = (const float*)d_in[0];
  const float* gamma  = (const float*)d_in[1];
  const float* beta   = (const float*)d_in[2];
  const float* w_qkv  = (const float*)d_in[3];
  const float* b_qkv  = (const float*)d_in[4];
  const float* w_proj = (const float*)d_in[5];
  const float* b_proj = (const float*)d_in[6];
  float* out = (float*)d_out;

  // workspace carve-up (all 1KB aligned)
  char* ws = (char*)d_ws;
  float* stats = (float*)ws;                                   // 64 f32
  unsigned short* wq = (unsigned short*)(ws + 1024);           // 1536*512 bf16
  unsigned short* wp = wq + 3 * C_ * C_;                       // 512*512 bf16
  unsigned short* ht = wp + C_ * C_;                           // [B][N][C] bf16
  unsigned short* qt = ht + (size_t)B_ * N_ * C_;              // [B][N][C] bf16 (scaled)
  unsigned short* kt = qt + (size_t)B_ * N_ * C_;              // [B][N][C] bf16
  unsigned short* vc = kt + (size_t)B_ * N_ * C_;              // [B][C][N] bf16
  unsigned short* at = vc + (size_t)B_ * N_ * C_;              // [B][N][C] bf16

  cvt_bf16<<<(3 * C_ * C_ + 255) / 256, 256, 0, stream>>>(w_qkv, wq, 3 * C_ * C_);
  cvt_bf16<<<(C_ * C_ + 255) / 256, 256, 0, stream>>>(w_proj, wp, C_ * C_);
  gn_stats<<<B_ * G_, 256, 0, stream>>>(x, stats);
  gn_apply<<<(B_ * C_ * N_) / 256, 256, 0, stream>>>(x, gamma, beta, stats, ht);
  qkv_gemm<<<dim3(N_ / 128, (3 * C_) / 128, B_), 256, 0, stream>>>(wq, ht, b_qkv, qt, kt, vc);
  attn_kernel<<<dim3(N_ / BR, B_), 256, 0, stream>>>(qt, kt, vc, at);
  proj_gemm<<<dim3(N_ / 128, C_ / 128, B_), 256, 0, stream>>>(wp, at, b_proj, x, out);
}
